// Cost_volume_62062277427554
// MI455X (gfx1250) — matrix operand order repaired and checked by reading, not compiled, not run on hardware
//
#include <hip/hip_runtime.h>
#include <math.h>

// Problem constants (from the reference)
#define B_   4
#define N_   4096
#define CIN_ 64
#define KQ_  16
#define KN_  4
#define BN_  (B_ * N_)      // 16384
#define M1_  (BN_ * KQ_)    // 262144 rows, stage-1 GEMMs
#define M2_  (BN_ * KN_)    // 65536 rows, stage-2 GEMMs

// Use CDNA5 async global->LDS copies for weight staging
#define USE_ASYNC_LDS 1

typedef __attribute__((ext_vector_type(16))) _Float16 v16h;
typedef __attribute__((ext_vector_type(8)))  float    v8f;

// --- WMMA fragment index maps (CDNA5 ISA 7.12.2, wave32) ---
// A (16x32 f16): lane L: M = L%16, half = L/16; VGPR v holds 2 f16:
//   v<4 : K = 2v + 8*half ; v>=4 : K = 16 + 2(v-4) + 8*half
__device__ __forceinline__ int a_kmap(int v, int hv) {
  return (v < 4) ? (v * 2 + hv * 8) : (16 + (v - 4) * 2 + hv * 8);
}
// B (32x16 f16): lane L: N = L%16; same K map as A
__device__ __forceinline__ int b_kmap(int v, int hv) {
  return (v < 4) ? (v * 2 + hv * 8) : (16 + (v - 4) * 2 + hv * 8);
}

// ============================================================
// Brute-force KNN: one thread per query, candidates staged in LDS
// ============================================================
template <int K>
__global__ void knn_kernel(const float* __restrict__ query,
                           const float* __restrict__ cand,
                           int* __restrict__ idx) {
  __shared__ float sx[1024], sy[1024], sz[1024];
  const int b  = blockIdx.y;
  const int qi = blockIdx.x * blockDim.x + threadIdx.x;
  const float qx = query[(b * N_ + qi) * 3 + 0];
  const float qy = query[(b * N_ + qi) * 3 + 1];
  const float qz = query[(b * N_ + qi) * 3 + 2];
  float bd[K]; int bi[K];
#pragma unroll
  for (int i = 0; i < K; ++i) { bd[i] = 3.4e38f; bi[i] = 0; }
  for (int t0 = 0; t0 < N_; t0 += 1024) {
    for (int t = threadIdx.x; t < 1024; t += blockDim.x) {
      const int ci = (b * N_ + t0 + t) * 3;
      sx[t] = cand[ci + 0]; sy[t] = cand[ci + 1]; sz[t] = cand[ci + 2];
    }
    __syncthreads();
    for (int t = 0; t < 1024; ++t) {
      const float dx = sx[t] - qx, dy = sy[t] - qy, dz = sz[t] - qz;
      const float d = dx * dx + dy * dy + dz * dz;
      if (d < bd[K - 1]) {
        int j = K - 1;
        while (j > 0 && bd[j - 1] > d) { bd[j] = bd[j - 1]; bi[j] = bi[j - 1]; --j; }
        bd[j] = d; bi[j] = t0 + t;
      }
    }
    __syncthreads();
  }
#pragma unroll
  for (int i = 0; i < K; ++i) idx[(b * N_ + qi) * K + i] = bi[i];
}

// ============================================================
// Weight pack: W[Cout,Cin] f32 -> B-fragments f16 [nt][kt][lane][16]
// ============================================================
__global__ void pack_w_kernel(const float* __restrict__ W, _Float16* __restrict__ frag,
                              int Cout, int Cin, int Ktiles) {
  const int nt = blockIdx.x, kt = blockIdx.y, lane = threadIdx.x;
  const int hv = lane >> 4;
  const int n  = nt * 16 + (lane & 15);
  _Float16* dst = frag + ((size_t)(nt * Ktiles + kt) * 32 + lane) * 16;
#pragma unroll
  for (int v = 0; v < 8; ++v)
#pragma unroll
    for (int p = 0; p < 2; ++p) {
      const int k = kt * 32 + b_kmap(v, hv) + p;
      float val = (n < Cout && k < Cin) ? W[(size_t)n * Cin + k] : 0.f;
      dst[v * 2 + p] = (_Float16)val;
    }
}

// ============================================================
// WMMA GEMM: one wave computes a full 16 x (NT*16) output strip.
// Weight fragments staged block-wide in LDS via async global->LDS copies.
// Per K-step: for each group of 4 N-tiles, batch the 8 ds_load_b128 of the
// B fragments then fire the 4-wmma burst -- keeps live B regs at 32 VGPRs
// (no scratch spills) while still overlapping LDS latency with XDL issue.
// A fragments software-pipelined from global. fp32 accum; bias +
// per-channel sum/sumsq epilogue (training-mode BN) fused.
// ============================================================
template <int NT>
__global__ void __launch_bounds__(256, 2)
wmma_gemm_kernel(const _Float16* __restrict__ Afrag,
                 const _Float16* __restrict__ Bfrag,
                 const float* __restrict__ bias,
                 float* __restrict__ out,
                 float* __restrict__ s_sum, float* __restrict__ s_sq,
                 int Ktiles, int Cout) {
  __shared__ __align__(32) _Float16 sB[NT * 6 * 512];  // up to 48 KB (NT=8, Kt=6)

  // ---- cooperative stage of B fragments into LDS ----
  const int total16 = NT * Ktiles * 64;  // 16-byte units (always multiple of 256)
  for (int u = threadIdx.x; u < total16; u += 256) {
#if USE_ASYNC_LDS
    const unsigned lds_off = (unsigned)(size_t)(&sB[0]) + (unsigned)u * 16u;
    const _Float16* g = Bfrag + (size_t)u * 8;
    asm volatile("global_load_async_to_lds_b128 %0, %1, off"
                 :: "v"(lds_off), "v"(g) : "memory");
#else
    ((uint4*)sB)[u] = ((const uint4*)Bfrag)[u];
#endif
  }
#if USE_ASYNC_LDS
  asm volatile("s_wait_asynccnt 0x0" ::: "memory");
#endif
  __syncthreads();

  const int wave = (blockIdx.x * blockDim.x + threadIdx.x) >> 5;  // == mtile
  const int lane = threadIdx.x & 31;
  const int mt = wave;

  v8f acc[NT];
#pragma unroll
  for (int nt = 0; nt < NT; ++nt) acc[nt] = (v8f){};

  const v16h* Ab = (const v16h*)Afrag + (size_t)mt * Ktiles * 32 + lane;
  const v16h* Bs = (const v16h*)sB + lane;

  v16h a = Ab[0];
  for (int kt = 0; kt < Ktiles; ++kt) {
    v16h a_next = a;
    if (kt + 1 < Ktiles) {
      __builtin_prefetch((const void*)(Ab + (kt + 2) * 32), 0, 3);
      a_next = Ab[(kt + 1) * 32];
    }
#pragma unroll
    for (int g = 0; g < NT; g += 4) {
      // batch 4 B-fragment LDS loads, then a 4-wmma burst
      v16h b0 = Bs[((g + 0) * Ktiles + kt) * 32];
      v16h b1 = Bs[((g + 1) * Ktiles + kt) * 32];
      v16h b2 = Bs[((g + 2) * Ktiles + kt) * 32];
      v16h b3 = Bs[((g + 3) * Ktiles + kt) * 32];
      acc[g + 0] = __builtin_amdgcn_wmma_f32_16x16x32_f16(false, a, false, b0,
                                                          (short)0, acc[g + 0],
                                                          false, false);
      acc[g + 1] = __builtin_amdgcn_wmma_f32_16x16x32_f16(false, a, false, b1,
                                                          (short)0, acc[g + 1],
                                                          false, false);
      acc[g + 2] = __builtin_amdgcn_wmma_f32_16x16x32_f16(false, a, false, b2,
                                                          (short)0, acc[g + 2],
                                                          false, false);
      acc[g + 3] = __builtin_amdgcn_wmma_f32_16x16x32_f16(false, a, false, b3,
                                                          (short)0, acc[g + 3],
                                                          false, false);
    }
    a = a_next;
  }

  // C layout: VGPR r, lanes 0-15 -> M=r, lanes 16-31 -> M=r+8; N = lane%16
  const int col0  = lane & 15;
  const int mbase = mt * 16 + ((lane >> 4) << 3);
#pragma unroll
  for (int nt = 0; nt < NT; ++nt) {
    const int ncol = nt * 16 + col0;
    const float bs = bias[ncol];
    float psum = 0.f, psq = 0.f;
#pragma unroll
    for (int r = 0; r < 8; ++r) {
      const float v = acc[nt][r] + bs;
      out[(size_t)(mbase + r) * Cout + ncol] = v;
      psum += v; psq += v * v;
    }
    atomicAdd(&s_sum[ncol], psum);
    atomicAdd(&s_sq[ncol],  psq);
  }
}

// Fold BN stats + gamma/beta into per-channel (a,b): y = relu(x*a + b)
__global__ void bn_finalize_kernel(const float* __restrict__ s_sum,
                                   const float* __restrict__ s_sq,
                                   const float* __restrict__ gamma,
                                   const float* __restrict__ beta,
                                   float* __restrict__ ab, int C, float invM) {
  const int c = threadIdx.x;
  if (c >= C) return;
  const float m = s_sum[c] * invM;
  const float v = s_sq[c] * invM - m * m;
  const float a = gamma[c] * rsqrtf(v + 1e-5f);
  ab[c]     = a;
  ab[C + c] = beta[c] - m * a;
}

// Elementwise BN+ReLU to fp32 (kept for softmax / weighted-sum consumers)
__global__ void bn_relu_apply_kernel(const float* __restrict__ X,
                                     const float* __restrict__ ab,
                                     float* __restrict__ Y, size_t total, int C) {
  const size_t i = blockIdx.x * (size_t)blockDim.x + threadIdx.x;
  if (i >= total) return;
  const int c = (int)(i % C);
  Y[i] = fmaxf(X[i] * ab[c] + ab[C + c], 0.f);
}

// ============================================================
// Activation packers: fp32 -> f16 A-fragments (one wave per (mtile,ktile))
// ============================================================
__global__ void pack_act_kernel(const float* __restrict__ X, const float* __restrict__ ab,
                                _Float16* __restrict__ frag, int Mrows, int Cin, int Ktiles) {
  const int wave = (blockIdx.x * blockDim.x + threadIdx.x) >> 5;
  const int lane = threadIdx.x & 31;
  const int mt = wave / Ktiles, kt = wave % Ktiles;
  if (mt >= (Mrows >> 4)) return;
  const int hv = lane >> 4;
  const int row = mt * 16 + (lane & 15);
  _Float16* dst = frag + ((size_t)(mt * Ktiles + kt) * 32 + lane) * 16;
#pragma unroll
  for (int v = 0; v < 8; ++v)
#pragma unroll
    for (int p = 0; p < 2; ++p) {
      const int k = kt * 32 + a_kmap(v, hv) + p;
      float val = 0.f;
      if (k < Cin) {
        val = X[(size_t)row * Cin + k];
        val = fmaxf(val * ab[k] + ab[Cin + k], 0.f);
      }
      dst[v * 2 + p] = (_Float16)val;
    }
}

// Stage-1 input: 138 channels = [pi_xyz, qi_xyz, diff, euc, pi_pts, qi_pts]
__global__ void pack_h0_kernel(const float* __restrict__ wxyz, const float* __restrict__ f2xyz,
                               const float* __restrict__ wpts, const float* __restrict__ f2pts,
                               const int* __restrict__ idxq, _Float16* __restrict__ frag,
                               int Ktiles) {
  const int wave = (blockIdx.x * blockDim.x + threadIdx.x) >> 5;
  const int lane = threadIdx.x & 31;
  const int mt = wave / Ktiles, kt = wave % Ktiles;
  if (mt >= (M1_ >> 4)) return;
  const int hv = lane >> 4;
  const int row = mt * 16 + (lane & 15);          // row = (b*N + n)*KQ + kq
  const int bn = row / KQ_;
  const int b  = bn / N_;
  const int q  = idxq[row];
  const int qg = b * N_ + q;
  const float px = wxyz[bn * 3 + 0], py = wxyz[bn * 3 + 1], pz = wxyz[bn * 3 + 2];
  const float qx = f2xyz[qg * 3 + 0], qy = f2xyz[qg * 3 + 1], qz = f2xyz[qg * 3 + 2];
  const float dx = qx - px, dy = qy - py, dz = qz - pz;
  const float euc = sqrtf(dx * dx + dy * dy + dz * dz + 1e-20f);
  _Float16* dst = frag + ((size_t)(mt * Ktiles + kt) * 32 + lane) * 16;
#pragma unroll
  for (int v = 0; v < 8; ++v)
#pragma unroll
    for (int p = 0; p < 2; ++p) {
      const int k = kt * 32 + a_kmap(v, hv) + p;
      float val = 0.f;
      if      (k == 0) val = px; else if (k == 1) val = py; else if (k == 2) val = pz;
      else if (k == 3) val = qx; else if (k == 4) val = qy; else if (k == 5) val = qz;
      else if (k == 6) val = dx; else if (k == 7) val = dy; else if (k == 8) val = dz;
      else if (k == 9) val = euc;
      else if (k < 10 + CIN_)     val = wpts[(size_t)bn * CIN_ + (k - 10)];
      else if (k < 10 + 2 * CIN_) val = f2pts[(size_t)qg * CIN_ + (k - 10 - CIN_)];
      dst[v * 2 + p] = (_Float16)val;
    }
}

// xyz_cat (10 channels) for enc1 / enc2 (Ktiles == 1)
__global__ void pack_xyzcat_kernel(const float* __restrict__ qxyz,
                                   const float* __restrict__ cxyz,
                                   const int* __restrict__ idx,
                                   _Float16* __restrict__ frag, int Mrows, int Kk) {
  const int wave = (blockIdx.x * blockDim.x + threadIdx.x) >> 5;
  const int lane = threadIdx.x & 31;
  if (wave >= (Mrows >> 4)) return;
  const int hv = lane >> 4;
  const int row = wave * 16 + (lane & 15);
  const int bn = row / Kk;
  const int b  = bn / N_;
  const int cg = b * N_ + idx[row];
  const float px = qxyz[bn * 3 + 0], py = qxyz[bn * 3 + 1], pz = qxyz[bn * 3 + 2];
  const float cx = cxyz[cg * 3 + 0], cy = cxyz[cg * 3 + 1], cz = cxyz[cg * 3 + 2];
  const float dx = cx - px, dy = cy - py, dz = cz - pz;
  const float euc = sqrtf(dx * dx + dy * dy + dz * dz + 1e-20f);
  const float ch[10] = {px, py, pz, cx, cy, cz, dx, dy, dz, euc};
  _Float16* dst = frag + ((size_t)wave * 32 + lane) * 16;
#pragma unroll
  for (int v = 0; v < 8; ++v)
#pragma unroll
    for (int p = 0; p < 2; ++p) {
      const int k = a_kmap(v, hv) + p;
      dst[v * 2 + p] = (_Float16)((k < 10) ? ch[k] : 0.f);
    }
}

// concat(encBN 64 || hBN 64) -> 128ch (mlp2 input)
__global__ void pack_cat2_kernel(const float* __restrict__ X0, const float* __restrict__ X1,
                                 _Float16* __restrict__ frag, int Mrows) {
  const int Ktiles = 4;
  const int wave = (blockIdx.x * blockDim.x + threadIdx.x) >> 5;
  const int lane = threadIdx.x & 31;
  const int mt = wave / Ktiles, kt = wave % Ktiles;
  if (mt >= (Mrows >> 4)) return;
  const int hv = lane >> 4;
  const int row = mt * 16 + (lane & 15);
  _Float16* dst = frag + ((size_t)(mt * Ktiles + kt) * 32 + lane) * 16;
#pragma unroll
  for (int v = 0; v < 8; ++v)
#pragma unroll
    for (int p = 0; p < 2; ++p) {
      const int k = kt * 32 + a_kmap(v, hv) + p;
      const float val = (k < 64) ? X0[(size_t)row * 64 + k]
                                 : X1[(size_t)row * 64 + (k - 64)];
      dst[v * 2 + p] = (_Float16)val;
    }
}

// concat(enc2BN 64 || warped_points 64 || gathered feat1 64) -> 192ch (mlp3 input)
__global__ void pack_cat3_kernel(const float* __restrict__ encBN,
                                 const float* __restrict__ wpts,
                                 const float* __restrict__ feat1,
                                 const int* __restrict__ idx2,
                                 _Float16* __restrict__ frag, int Mrows) {
  const int Ktiles = 6;
  const int wave = (blockIdx.x * blockDim.x + threadIdx.x) >> 5;
  const int lane = threadIdx.x & 31;
  const int mt = wave / Ktiles, kt = wave % Ktiles;
  if (mt >= (Mrows >> 4)) return;
  const int hv = lane >> 4;
  const int row = mt * 16 + (lane & 15);        // row = bn*KN + kn
  const int bn = row / KN_;
  const int b  = bn / N_;
  const int nb = idx2[row];
  _Float16* dst = frag + ((size_t)(mt * Ktiles + kt) * 32 + lane) * 16;
#pragma unroll
  for (int v = 0; v < 8; ++v)
#pragma unroll
    for (int p = 0; p < 2; ++p) {
      const int k = kt * 32 + a_kmap(v, hv) + p;
      float val = 0.f;
      if      (k < 64)  val = encBN[(size_t)row * 64 + k];
      else if (k < 128) val = wpts[(size_t)bn * 64 + (k - 64)];
      else if (k < 192) val = feat1[((size_t)b * N_ + nb) * 64 + (k - 128)];
      dst[v * 2 + p] = (_Float16)val;
    }
}

// ============================================================
// Softmax over neighbors + weighted sum
// ============================================================
__global__ void softmax_ws_kernel(const float* __restrict__ P, const float* __restrict__ H,
                                  float* __restrict__ out, int K, int C) {
  const int i = blockIdx.x * blockDim.x + threadIdx.x;
  if (i >= BN_ * C) return;
  const int bn = i / C, c = i % C;
  float mx = -3.4e38f;
  for (int k = 0; k < K; ++k) mx = fmaxf(mx, P[((size_t)bn * K + k) * C + c]);
  float s = 0.f, acc = 0.f;
  for (int k = 0; k < K; ++k) {
    const float e = __expf(P[((size_t)bn * K + k) * C + c] - mx);
    s += e; acc += e * H[((size_t)bn * K + k) * C + c];
  }
  out[(size_t)bn * C + c] = acc / s;
}

__global__ void softmax_ws_gather_kernel(const float* __restrict__ P,
                                         const float* __restrict__ feat,
                                         const int* __restrict__ idx,
                                         float* __restrict__ out, int K, int C) {
  const int i = blockIdx.x * blockDim.x + threadIdx.x;
  if (i >= BN_ * C) return;
  const int bn = i / C, c = i % C;
  const int b = bn / N_;
  float mx = -3.4e38f;
  for (int k = 0; k < K; ++k) mx = fmaxf(mx, P[((size_t)bn * K + k) * C + c]);
  float s = 0.f, acc = 0.f;
  for (int k = 0; k < K; ++k) {
    const float e = __expf(P[((size_t)bn * K + k) * C + c] - mx);
    const int nb = idx[bn * K + k];
    s += e; acc += e * feat[((size_t)b * N_ + nb) * C + c];
  }
  out[(size_t)bn * C + c] = acc / s;
}

// ============================================================
// Host orchestration
// ============================================================
extern "C" void kernel_launch(void* const* d_in, const int* in_sizes, int n_in,
                              void* d_out, int out_size, void* d_ws, size_t ws_size,
                              hipStream_t stream) {
  (void)in_sizes; (void)n_in; (void)out_size; (void)ws_size;
  auto F = [&](int i) { return (const float*)d_in[i]; };
  const float* wxyz  = F(0);
  const float* f2xyz = F(1);
  const float* wpts  = F(2);
  const float* f2pts = F(3);
  // mlp1: idx 4..15, enc1: 16..19, mlp2: 20..27, enc2: 28..31, mlp3: 32..35

  // ---- workspace carve ----
  size_t off = 0;
  auto carve = [&](size_t bytes) -> char* {
    char* p = (char*)d_ws + off;
    off += (bytes + 255) & ~(size_t)255;
    return p;
  };
  int*      idxq  = (int*)carve((size_t)M1_ * 4);
  int*      idx2  = (int*)carve((size_t)M2_ * 4);
  _Float16* frag  = (_Float16*)carve((size_t)(M1_ / 16) * 5 * 1024);  // max A-frag
  _Float16* wf[9];
  for (int i = 0; i < 9; ++i) wf[i] = (_Float16*)carve(64 * 1024);
  float* ping  = (float*)carve((size_t)M1_ * 128 * 4);
  float* pong  = (float*)carve((size_t)M1_ * 64 * 4);
  float* hBN   = (float*)carve((size_t)M1_ * 64 * 4);
  float* eBN   = (float*)carve((size_t)M1_ * 64 * 4);   // reused as pcBN
  float* feat1 = (float*)carve((size_t)BN_ * 64 * 4);
  float* e2BN  = (float*)carve((size_t)M2_ * 64 * 4);   // reused as xBN
  float* stats = (float*)carve(9 * 512 * 4);            // [layer]{sum128|sq128|ab256}

  hipMemsetAsync(stats, 0, 9 * 512 * 4, stream);

  // ---- helpers ----
  auto packW = [&](int widx, _Float16* dst, int Cout, int Cin) {
    const int Nt = Cout / 16, Kt = (Cin + 31) / 32;
    pack_w_kernel<<<dim3(Nt, Kt), 32, 0, stream>>>(F(widx), dst, Cout, Cin, Kt);
  };
  auto gemm = [&](const _Float16* A, const _Float16* Bf, int bidx, float* out,
                  int Mtiles, int Cout, int Ktiles, float* st) {
    const int blocks = Mtiles / 8;  // one wave per mtile, 8 waves per block
    if (Cout == 128)
      wmma_gemm_kernel<8><<<blocks, 256, 0, stream>>>(A, Bf, F(bidx), out, st,
                                                      st + 128, Ktiles, Cout);
    else
      wmma_gemm_kernel<4><<<blocks, 256, 0, stream>>>(A, Bf, F(bidx), out, st,
                                                      st + 128, Ktiles, Cout);
  };
  auto finalize = [&](float* st, int gidx, int bidx, int C, float invM) {
    bn_finalize_kernel<<<1, 128, 0, stream>>>(st, st + 128, F(gidx), F(bidx),
                                              st + 256, C, invM);
  };
  auto packA = [&](const float* X, const float* ab, int Mrows, int Cin) {
    const int waves = (Mrows / 16) * ((Cin + 31) / 32);
    pack_act_kernel<<<(waves + 7) / 8, 256, 0, stream>>>(X, ab, frag, Mrows, Cin,
                                                         (Cin + 31) / 32);
  };
  auto bnApply = [&](const float* X, const float* ab, float* Y, size_t M, int C) {
    const size_t total = M * C;
    bn_relu_apply_kernel<<<(unsigned)((total + 255) / 256), 256, 0, stream>>>(
        X, ab, Y, total, C);
  };
  const float inv1 = 1.f / (float)M1_, inv2 = 1.f / (float)M2_;

  // ---- weight packs (cheap; re-run every call for determinism) ----
  packW(4,  wf[0], 128, 138);   // mlp1 L0
  packW(8,  wf[1], 64, 128);    // mlp1 L1
  packW(12, wf[2], 64, 64);     // mlp1 L2
  packW(16, wf[3], 64, 10);     // enc1
  packW(20, wf[4], 128, 128);   // mlp2 L0
  packW(24, wf[5], 64, 128);    // mlp2 L1
  packW(28, wf[6], 64, 10);     // enc2
  packW(32, wf[7], 128, 192);   // mlp3 L0
  packW(34, wf[8], 64, 128);    // mlp3 L1

  // ---- stage 1 ----
  knn_kernel<KQ_><<<dim3(N_ / 256, B_), 256, 0, stream>>>(wxyz, f2xyz, idxq);

  float* st = stats + 0 * 512;  // mlp1 L0
  pack_h0_kernel<<<((M1_ / 16) * 5 + 7) / 8, 256, 0, stream>>>(wxyz, f2xyz, wpts,
                                                               f2pts, idxq, frag, 5);
  gemm(frag, wf[0], 5, ping, M1_ / 16, 128, 5, st);
  finalize(st, 6, 7, 128, inv1);

  packA(ping, st + 256, M1_, 128);          // mlp1 L1
  st = stats + 1 * 512;
  gemm(frag, wf[1], 9, pong, M1_ / 16, 64, 4, st);
  finalize(st, 10, 11, 64, inv1);

  packA(pong, st + 256, M1_, 64);           // mlp1 L2
  st = stats + 2 * 512;
  gemm(frag, wf[2], 13, ping, M1_ / 16, 64, 2, st);
  finalize(st, 14, 15, 64, inv1);
  bnApply(ping, st + 256, hBN, M1_, 64);    // h (post BN+ReLU)

  // enc1
  pack_xyzcat_kernel<<<((M1_ / 16) + 7) / 8, 256, 0, stream>>>(wxyz, f2xyz, idxq,
                                                               frag, M1_, KQ_);
  st = stats + 3 * 512;
  gemm(frag, wf[3], 17, pong, M1_ / 16, 64, 1, st);
  finalize(st, 18, 19, 64, inv1);
  bnApply(pong, st + 256, eBN, M1_, 64);

  // mlp2 L0
  pack_cat2_kernel<<<((M1_ / 16) * 4 + 7) / 8, 256, 0, stream>>>(eBN, hBN, frag, M1_);
  st = stats + 4 * 512;
  gemm(frag, wf[4], 21, ping, M1_ / 16, 128, 4, st);
  finalize(st, 22, 23, 128, inv1);

  packA(ping, st + 256, M1_, 128);          // mlp2 L1
  st = stats + 5 * 512;
  gemm(frag, wf[5], 25, pong, M1_ / 16, 64, 4, st);
  finalize(st, 26, 27, 64, inv1);
  bnApply(pong, st + 256, eBN, M1_, 64);    // pcBN (aliases eBN, already consumed)

  softmax_ws_kernel<<<(BN_ * 64 + 255) / 256, 256, 0, stream>>>(eBN, hBN, feat1,
                                                                KQ_, 64);

  // ---- stage 2 ----
  knn_kernel<KN_><<<dim3(N_ / 256, B_), 256, 0, stream>>>(wxyz, wxyz, idx2);

  pack_xyzcat_kernel<<<((M2_ / 16) + 7) / 8, 256, 0, stream>>>(wxyz, wxyz, idx2,
                                                               frag, M2_, KN_);
  st = stats + 6 * 512;                     // enc2
  gemm(frag, wf[6], 29, pong, M2_ / 16, 64, 1, st);
  finalize(st, 30, 31, 64, inv2);
  bnApply(pong, st + 256, e2BN, M2_, 64);

  // mlp3 L0 (BN gamma/beta reused from mlp2 L0, fresh stats)
  pack_cat3_kernel<<<((M2_ / 16) * 6 + 7) / 8, 256, 0, stream>>>(e2BN, wpts, feat1,
                                                                 idx2, frag, M2_);
  st = stats + 7 * 512;
  gemm(frag, wf[7], 33, ping, M2_ / 16, 128, 6, st);
  finalize(st, 22, 23, 128, inv2);

  packA(ping, st + 256, M2_, 128);          // mlp3 L1 (BN from mlp2 L1)
  st = stats + 8 * 512;
  gemm(frag, wf[8], 35, pong, M2_ / 16, 64, 4, st);
  finalize(st, 26, 27, 64, inv2);
  bnApply(pong, st + 256, e2BN, M2_, 64);   // xBN (aliases e2BN, already consumed)

  softmax_ws_gather_kernel<<<(BN_ * 64 + 255) / 256, 256, 0, stream>>>(
      e2BN, feat1, idx2, (float*)d_out, KN_, 64);
}
